// RNNBase_37950331027581
// MI455X (gfx1250) — compile-verified, matrix operand order checked by reading, not run on hardware
//
#include <hip/hip_runtime.h>
#include <hip/hip_bf16.h>
#include <hip/hip_fp16.h>

// ---------------- problem constants ----------------
#define SEQ     512
#define BATCH   64
#define IN_DIM  1024
#define HID     1024
#define G4      4096      // 4*H
#define KTOT    2048      // IN + H  (fused [W_ih | W_hh] K dimension)
#define QSCALE  0.1f

#define NBLK    128       // persistent grid: 128 WGs x 256 thr = 1024 wave32
#define NTHR    256
#define NLANES  (NBLK * NTHR)   // 32768

typedef __attribute__((ext_vector_type(16))) _Float16 v16h;
typedef __attribute__((ext_vector_type(8)))  _Float16 v8h;
typedef __attribute__((ext_vector_type(8)))  float    v8f;

// ---------------------------------------------------------------------------
// Fragment loader for 16-bit A/B operands of V_WMMA_F32_16X16X32_F16.
// ISA 7.12.2 (16-bit A 16x32): lanes 0-15 hold row M=lane, K = kb+{0..7,16..23};
// lanes 16-31 hold row M=lane-16, K = kb+{8..15,24..31}.  B (32x16, col-major
// source = rows of the [N,K] weight array) uses the mirrored pattern with the
// lane's low 4 bits selecting the column.
// ---------------------------------------------------------------------------
__device__ inline v16h load_frag16(const _Float16* __restrict__ base, int ld,
                                   int lane, int kb) {
  const int r  = lane & 15;
  const int ko = (lane >> 4) << 3;           // 0 or 8
  const _Float16* p = base + (size_t)r * ld + kb + ko;
  v8h lo = *(const v8h*)(p);                 // K = kb+ko   .. kb+ko+7
  v8h hi = *(const v8h*)(p + 16);            // K = kb+ko+16.. kb+ko+23
  return __builtin_shufflevector(lo, hi, 0,1,2,3,4,5,6,7,8,9,10,11,12,13,14,15);
}

// ---------------------------------------------------------------------------
// Device-wide split barrier (arrive counter + generation), agent scope.
// ---------------------------------------------------------------------------
__device__ inline void grid_sync(unsigned* bar) {
  __threadfence();
  __syncthreads();
  if (threadIdx.x == 0) {
    unsigned gen = __hip_atomic_load(bar + 1, __ATOMIC_ACQUIRE, __HIP_MEMORY_SCOPE_AGENT);
    unsigned old = __hip_atomic_fetch_add(bar, 1u, __ATOMIC_ACQ_REL, __HIP_MEMORY_SCOPE_AGENT);
    if (old == NBLK - 1) {
      __hip_atomic_store(bar, 0u, __ATOMIC_RELAXED, __HIP_MEMORY_SCOPE_AGENT);
      __hip_atomic_fetch_add(bar + 1, 1u, __ATOMIC_ACQ_REL, __HIP_MEMORY_SCOPE_AGENT);
    } else {
      while (__hip_atomic_load(bar + 1, __ATOMIC_ACQUIRE, __HIP_MEMORY_SCOPE_AGENT) == gen)
        __builtin_amdgcn_s_sleep(1);
    }
  }
  __syncthreads();
}

__device__ inline float sigmoid_f(float x) {
  return 1.0f / (1.0f + __expf(-x));
}

// ---------------------------------------------------------------------------
// Prep kernels
// ---------------------------------------------------------------------------
__global__ __launch_bounds__(NTHR) void k_cvt_x(const float* __restrict__ x,
                                                _Float16* __restrict__ x16) {
  int idx = blockIdx.x * NTHR + threadIdx.x;    // 33,554,432 exact
  x16[idx] = (_Float16)x[idx];
}

// fake-quant (qint8, scale 0.1, zp 0) both weight matrices into fused f16 [4H, IN+H]
__global__ __launch_bounds__(NTHR) void k_quant_w(const float* __restrict__ w_ih,
                                                  const float* __restrict__ w_hh,
                                                  _Float16* __restrict__ wcat) {
  int idx = blockIdx.x * NTHR + threadIdx.x;    // 8,388,608 exact
  int n = idx >> 11;          // row in [0, 4096)
  int k = idx & (KTOT - 1);   // col in [0, 2048)
  float w = (k < IN_DIM) ? w_ih[(size_t)n * IN_DIM + k]
                         : w_hh[(size_t)n * HID + (k - IN_DIM)];
  float q = fminf(fmaxf(rintf(w * (1.0f / QSCALE)), -128.0f), 127.0f) * QSCALE;
  wcat[idx] = (_Float16)q;
}

__global__ __launch_bounds__(NTHR) void k_init(const float* __restrict__ h0,
                                               const float* __restrict__ c0,
                                               const float* __restrict__ b_ih,
                                               const float* __restrict__ b_hh,
                                               _Float16* __restrict__ h16,
                                               float* __restrict__ cst,
                                               float* __restrict__ bias,
                                               unsigned* __restrict__ bar) {
  int idx = blockIdx.x * NTHR + threadIdx.x;    // 65,536 exact
  h16[idx] = (_Float16)h0[idx];
  cst[idx] = c0[idx];
  if (idx < G4) bias[idx] = b_ih[idx] + b_hh[idx];
  if (idx == 0) bar[0] = 0u;                    // arrive counter must start at 0
}

// ---------------------------------------------------------------------------
// Persistent LSTM kernel: one wave owns one 16x16 tile of gates [64 x 4096].
// Per step: gates = [x_t , h] * [W_ih ; W_hh]^T + bias, activated per gate,
// then 2 cells/thread of the c/h elementwise update.  Two grid barriers/step.
// ---------------------------------------------------------------------------
__global__ __launch_bounds__(NTHR) void lstm_persist(
    const _Float16* __restrict__ x16,    // [SEQ][B][IN] f16
    const _Float16* __restrict__ wcat,   // [4H][KTOT]  f16 (quantized)
    const float*    __restrict__ bias,   // [4H]
    _Float16*       __restrict__ h16,    // [B][H] f16  (A operand of recurrence)
    float*          __restrict__ cst,    // [B][H] f32  cell state
    float*          __restrict__ gates,  // [B][4H] f32 activated gates
    unsigned*       __restrict__ bar,    // barrier {count, gen}
    float*          __restrict__ out)    // [SEQ][B][H] ++ h_n[B][H] ++ c_n[B][H]
{
  const int tid  = blockIdx.x * NTHR + threadIdx.x;
  const int lane = threadIdx.x & 31;
  const int wid  = tid >> 5;            // 0..1023
  const int mt   = wid >> 8;            // row tile   0..3   (batch/16)
  const int nt   = wid & 255;           // col tile   0..255 (4H/16)
  const int gate = nt >> 6;             // 0:i 1:f 2:g 3:o
  const int ncol = nt * 16 + (lane & 15);
  const float bn = bias[ncol];
  const int rowhalf = (lane >> 4) << 3; // 0 or 8 : C-layout row offset
  const _Float16* bbase = wcat + (size_t)(nt * 16) * KTOT;
  const _Float16* abase_h = h16 + (size_t)(mt * 16) * HID;

  for (int t = 0; t < SEQ; ++t) {
    // ---- gate GEMM: K = 2048 (x part then h part), f16 WMMA, f32 acc ----
    const _Float16* abase_x = x16 + (size_t)t * (BATCH * IN_DIM) + (size_t)(mt * 16) * IN_DIM;
    v8f acc = {};
#pragma unroll 4
    for (int kc = 0; kc < 32; ++kc) {
      v16h a = load_frag16(abase_x, IN_DIM, lane, kc * 32);
      v16h b = load_frag16(bbase,   KTOT,   lane, kc * 32);
      acc = __builtin_amdgcn_wmma_f32_16x16x32_f16(false, a, false, b,
                                                   (short)0, acc, false, false);
    }
#pragma unroll 4
    for (int kc = 0; kc < 32; ++kc) {
      v16h a = load_frag16(abase_h, HID,  lane, kc * 32);
      v16h b = load_frag16(bbase,   KTOT, lane, IN_DIM + kc * 32);
      acc = __builtin_amdgcn_wmma_f32_16x16x32_f16(false, a, false, b,
                                                   (short)0, acc, false, false);
    }

    // ---- bias + activation + scatter to gates[B][4H] (C-layout store) ----
#pragma unroll
    for (int v = 0; v < 8; ++v) {
      float g = acc[v] + bn;
      g = (gate == 2) ? tanhf(g) : sigmoid_f(g);
      int m = mt * 16 + v + rowhalf;
      gates[(size_t)m * G4 + ncol] = g;
    }

    grid_sync(bar);   // all gates visible

    // ---- elementwise LSTM cell update: 65536 cells / 32768 threads ----
#pragma unroll
    for (int r = 0; r < 2; ++r) {
      int e = tid + r * NLANES;
      int b = e >> 10;
      int j = e & (HID - 1);
      const float* gr = gates + (size_t)b * G4;
      float ig = gr[j];
      float fg = gr[HID + j];
      float gg = gr[2 * HID + j];
      float og = gr[3 * HID + j];
      float cn = fg * cst[e] + ig * gg;
      float hn = og * tanhf(cn);
      cst[e] = cn;
      out[(size_t)t * (BATCH * HID) + e] = hn;
      h16[e] = (_Float16)hn;
      if (t == SEQ - 1) {
        out[(size_t)SEQ * (BATCH * HID) + e] = hn;                 // h_n
        out[(size_t)SEQ * (BATCH * HID) + (BATCH * HID) + e] = cn; // c_n
      }
    }

    // warm L0/L2 for next timestep's activations while we wait
    if (t + 1 < SEQ && tid < 1024) {
      const char* pf = (const char*)(x16 + (size_t)(t + 1) * (BATCH * IN_DIM));
      __builtin_prefetch(pf + tid * 128, 0, 0);
    }

    grid_sync(bar);   // h16 / cst updates visible before next GEMM
  }
}

// ---------------------------------------------------------------------------
// host-side launcher
// ---------------------------------------------------------------------------
extern "C" void kernel_launch(void* const* d_in, const int* in_sizes, int n_in,
                              void* d_out, int out_size, void* d_ws, size_t ws_size,
                              hipStream_t stream) {
  (void)in_sizes; (void)n_in; (void)out_size; (void)ws_size;
  const float* x    = (const float*)d_in[0];
  const float* h0   = (const float*)d_in[1];
  const float* c0   = (const float*)d_in[2];
  const float* w_ih = (const float*)d_in[3];
  const float* w_hh = (const float*)d_in[4];
  const float* b_ih = (const float*)d_in[5];
  const float* b_hh = (const float*)d_in[6];
  float* out = (float*)d_out;

  // workspace carve-up (all slices naturally 256B-aligned)
  char* w = (char*)d_ws;
  _Float16* x16  = (_Float16*)w;  w += (size_t)SEQ * BATCH * IN_DIM * sizeof(_Float16); // 64 MB
  _Float16* wcat = (_Float16*)w;  w += (size_t)G4 * KTOT * sizeof(_Float16);            // 16 MB
  float*    bias = (float*)w;     w += (size_t)G4 * sizeof(float);
  _Float16* h16  = (_Float16*)w;  w += (size_t)BATCH * HID * sizeof(_Float16);
  float*    cst  = (float*)w;     w += (size_t)BATCH * HID * sizeof(float);
  float*    gts  = (float*)w;     w += (size_t)BATCH * G4 * sizeof(float);
  unsigned* bar  = (unsigned*)w;

  k_cvt_x  <<<(SEQ * BATCH * IN_DIM) / NTHR, NTHR, 0, stream>>>(x, x16);
  k_quant_w<<<(G4 * KTOT) / NTHR,            NTHR, 0, stream>>>(w_ih, w_hh, wcat);
  k_init   <<<(BATCH * HID) / NTHR,          NTHR, 0, stream>>>(h0, c0, b_ih, b_hh,
                                                                h16, cst, bias, bar);
  lstm_persist<<<NBLK, NTHR, 0, stream>>>(x16, wcat, bias, h16, cst, gts, bar, out);
}
